// PhysicsForwardModel_1992864825659
// MI455X (gfx1250) — matrix operand order repaired and checked by reading, not compiled, not run on hardware
//
#include <hip/hip_runtime.h>
#include <hip/hip_bf16.h>
#include <math.h>

typedef __attribute__((ext_vector_type(16))) _Float16 v16h;
typedef __attribute__((ext_vector_type(8)))  float    v8f;

// ---- CDNA5 wave32 WMMA fragment index helpers (per CDNA5 ISA) ----
// A (16x32 f16): lane M = lane&15; element (vgpr v, pos p):
//   K = 16*(v>>2) + 8*(lane>=16) + 2*(v&3) + p
__device__ __forceinline__ int a_frag_k(int v, int p, int hi) {
    return ((v >> 2) << 4) + (hi << 3) + ((v & 3) << 1) + p;
}
// B (32x16 f16): lane N = lane&15; element (vgpr v, pos p):
//   K = 16*(v>>2) + 8*(lane>=16) + 2*(v&3) + p
__device__ __forceinline__ int b_frag_k(int v, int p, int hi) {
    return ((v >> 2) << 4) + (hi << 3) + ((v & 3) << 1) + p;
}
// C/D (16x16 f32): vgpr r -> M = r + 8*(lane>=16), N = lane&15

#define PI_F 3.14159265358979323846f

// ---------------------------------------------------------------------------
// K1: build DCT tables.
//   blocks 0..255   : Afull[512][128],  A[y][n] = s(y)*cos(pi*(2n+1)*y/1024)
//   blocks 256..511 : BmT[128][512],    BmT[m][x] = s(x)*cos(pi*(2*(m+128)+1)*x/1024)
//   blocks 512..513 : vvec[512],        v[y] = s(y)*sum_k s(k)*cos(pi*(2k+1)*y/1024)
// ---------------------------------------------------------------------------
__global__ void k1_tables(float* __restrict__ Afull, float* __restrict__ BmT,
                          float* __restrict__ vvec) {
    const int blk = blockIdx.x, tid = threadIdx.x;
    const float sc0 = sqrtf(1.0f / 512.0f), sc1 = sqrtf(2.0f / 512.0f);
    if (blk < 256) {
        int e = blk * 256 + tid;
        int y = e >> 7, n = e & 127;
        float s = (y == 0) ? sc0 : sc1;
        Afull[e] = s * cosf(PI_F * (2.0f * n + 1.0f) * (float)y * (1.0f / 1024.0f));
    } else if (blk < 512) {
        int e = (blk - 256) * 256 + tid;
        int m = e >> 9, xc = e & 511;
        float s = (xc == 0) ? sc0 : sc1;
        BmT[e] = s * cosf(PI_F * (2.0f * (m + 128) + 1.0f) * (float)xc * (1.0f / 1024.0f));
    } else {
        int y = (blk - 512) * 256 + tid;
        float s = (y == 0) ? sc0 : sc1;
        float acc = 0.0f;
        for (int k = 0; k < 512; ++k) {
            float sk = (k == 0) ? sc0 : sc1;
            acc += sk * cosf(PI_F * (2.0f * k + 1.0f) * (float)y * (1.0f / 1024.0f));
        }
        vvec[y] = s * acc;
    }
}

// ---------------------------------------------------------------------------
// K2: T[b][n][x] = sum_m x[b][n][m] * BmT[m][x]   (M=128, N=512, K=128, 4 batches)
// One wave per 16x16 tile; f16 WMMA, f32 accumulate.
// ---------------------------------------------------------------------------
__global__ void k2_xBmT(const float* __restrict__ x, const float* __restrict__ BmT,
                        float* __restrict__ T) {
    const int blk = blockIdx.x;                 // 4 * 8 * 32 = 1024
    const int b  = blk >> 8;
    const int mt = (blk >> 5) & 7;
    const int nt = blk & 31;
    const int lane = threadIdx.x;
    const int hi = lane >> 4, lm = lane & 15;
    const float* xb = x + b * 128 * 128;
    v8f acc = {};
    for (int ks = 0; ks < 4; ++ks) {
        v16h af = {}, bf = {};
        const int row = mt * 16 + lm;
#pragma unroll
        for (int v = 0; v < 8; ++v)
#pragma unroll
            for (int p = 0; p < 2; ++p) {
                int k = ks * 32 + a_frag_k(v, p, hi);
                af[2 * v + p] = (_Float16)xb[row * 128 + k];
            }
        const int col = nt * 16 + lm;
#pragma unroll
        for (int v = 0; v < 8; ++v)
#pragma unroll
            for (int p = 0; p < 2; ++p) {
                int k = ks * 32 + b_frag_k(v, p, hi);
                bf[2 * v + p] = (_Float16)BmT[k * 512 + col];
            }
        acc = __builtin_amdgcn_wmma_f32_16x16x32_f16(false, af, false, bf,
                                                     (short)0, acc, false, false);
    }
    float* Tb = T + b * 128 * 512;
#pragma unroll
    for (int r = 0; r < 8; ++r) {
        int row = mt * 16 + r + 8 * hi;
        Tb[row * 512 + (nt * 16 + lm)] = acc[r];
    }
}

// ---------------------------------------------------------------------------
// K3: Wt[b][j][i] = (f16)( v[i] * sum_n Afull[i][n] * T[b][n][j] )
//     (M=i=512, N=j=512, K=n=128, 4 batches) — TRANSPOSED store (j-major)
//     so K4 can fill LDS with coalesced reads of column j.
// ---------------------------------------------------------------------------
__global__ void k3_AT(const float* __restrict__ Afull, const float* __restrict__ T,
                      const float* __restrict__ vvec, _Float16* __restrict__ Wt) {
    const int blk = blockIdx.x;                 // 4 * 32 * 32 = 4096
    const int b  = blk >> 10;
    const int it = (blk >> 5) & 31;
    const int jt = blk & 31;
    const int lane = threadIdx.x;
    const int hi = lane >> 4, lm = lane & 15;
    const float* Tb = T + b * 128 * 512;
    v8f acc = {};
    for (int ks = 0; ks < 4; ++ks) {
        v16h af = {}, bf = {};
        const int row = it * 16 + lm;
#pragma unroll
        for (int v = 0; v < 8; ++v)
#pragma unroll
            for (int p = 0; p < 2; ++p) {
                int k = ks * 32 + a_frag_k(v, p, hi);
                af[2 * v + p] = (_Float16)Afull[row * 128 + k];
            }
        const int col = jt * 16 + lm;
#pragma unroll
        for (int v = 0; v < 8; ++v)
#pragma unroll
            for (int p = 0; p < 2; ++p) {
                int k = ks * 32 + b_frag_k(v, p, hi);
                bf[2 * v + p] = (_Float16)Tb[k * 512 + col];
            }
        acc = __builtin_amdgcn_wmma_f32_16x16x32_f16(false, af, false, bf,
                                                     (short)0, acc, false, false);
    }
    _Float16* Wb = Wt + b * 512 * 512;
#pragma unroll
    for (int r = 0; r < 8; ++r) {
        int i  = it * 16 + r + 8 * hi;
        int jc = jt * 16 + lm;
        Wb[jc * 512 + i] = (_Float16)(vvec[i] * acc[r]);
    }
}

// ---------------------------------------------------------------------------
// K4 (dominant): Pdet[b][t][j] = (1/3) * sum_i att(t,i,j) * W[b][i][j]
//   att generated in registers: exp(-chi*F*t)*cos(F*t), F=(pi/512)*sqrt(i^2+j^2)
//   One block per j column (512 blocks, 8 waves). W column + F column in LDS.
//   ks OUTER / t-tile INNER: the B fragment (W slice) and the 16 F values are
//   invariant across the 4 t-tiles of a wave, so they are loaded once per ks;
//   the 4 independent accumulator chains let WMMAs overlap the exp/cos VALU
//   generation of the next A fragment.
// ---------------------------------------------------------------------------
__global__ void k4_att(const _Float16* __restrict__ Wt, const float* __restrict__ chip,
                       float* __restrict__ out) {
    __shared__ float    sF[512];
    __shared__ _Float16 sW[4 * 512];
    const int j   = blockIdx.x;
    const int tid = threadIdx.x;
    const float chi = chip[0];
    const float kf  = PI_F / 512.0f;

    for (int i = tid; i < 512; i += 256) {
        float fi = (float)i, fj = (float)j;
        sF[i] = kf * sqrtf(fi * fi + fj * fj);
    }
    for (int e = tid; e < 2048; e += 256) {           // coalesced: Wt is j-major
        int b = e >> 9, i = e & 511;
        sW[e] = Wt[((size_t)b * 512 + j) * 512 + i];
    }
    __syncthreads();

    const int wave = tid >> 5, lane = tid & 31;
    const int hi = lane >> 4, lm = lane & 15;

    float tf[4];
#pragma unroll
    for (int tti = 0; tti < 4; ++tti)
        tf[tti] = (float)((wave + tti * 8) * 16 + lm);   // this lane's t per tile

    v8f acc[4] = {v8f{}, v8f{}, v8f{}, v8f{}};

    for (int ks = 0; ks < 16; ++ks) {
        // F values for this wave's A-fragment lanes: invariant across t-tiles
        float fr[16];
#pragma unroll
        for (int v = 0; v < 8; ++v)
#pragma unroll
            for (int p = 0; p < 2; ++p)
                fr[2 * v + p] = sF[ks * 32 + a_frag_k(v, p, hi)];

        // B fragment (W column slice): invariant across t-tiles
        v16h bf = {};
#pragma unroll
        for (int v = 0; v < 8; ++v)
#pragma unroll
            for (int p = 0; p < 2; ++p) {
                int i = ks * 32 + b_frag_k(v, p, hi);
                bf[2 * v + p] = (lm < 4) ? sW[lm * 512 + i] : (_Float16)0.0f;
            }

#pragma unroll
        for (int tti = 0; tti < 4; ++tti) {
            v16h af = {};
#pragma unroll
            for (int e = 0; e < 16; ++e) {
                float arg = fr[e] * tf[tti];
                af[e] = (_Float16)(__expf(-chi * arg) * __cosf(arg));
            }
            acc[tti] = __builtin_amdgcn_wmma_f32_16x16x32_f16(false, af, false, bf,
                                                              (short)0, acc[tti],
                                                              false, false);
        }
    }

    if (lm < 4) {                                      // only batches 0..3 are real
#pragma unroll
        for (int tti = 0; tti < 4; ++tti) {
            int tt = wave + tti * 8;
#pragma unroll
            for (int r = 0; r < 8; ++r) {
                int t = tt * 16 + r + 8 * hi;
                out[((size_t)lm * 512 + t) * 512 + j] = acc[tti][r] * (1.0f / 3.0f);
            }
        }
    }
}

// ---------------------------------------------------------------------------
extern "C" void kernel_launch(void* const* d_in, const int* in_sizes, int n_in,
                              void* d_out, int out_size, void* d_ws, size_t ws_size,
                              hipStream_t stream) {
    const float* x    = (const float*)d_in[0];   // (4,1,128,128) f32
    const float* chip = (const float*)d_in[1];   // scalar chi
    (void)in_sizes; (void)n_in; (void)out_size; (void)ws_size;
    float* out = (float*)d_out;                  // (4,1,512,512) f32

    char* ws = (char*)d_ws;                      // ~3.5 MB used
    float*     Afull = (float*)(ws + 0);               // 512*128 f32  = 256 KB
    float*     BmT   = (float*)(ws + 262144);          // 128*512 f32  = 256 KB
    float*     vvec  = (float*)(ws + 524288);          // 512 f32      = 2 KB
    float*     T     = (float*)(ws + 526336);          // 4*128*512 f32= 1 MB
    _Float16*  Wt    = (_Float16*)(ws + 1574912);      // 4*512*512 f16= 2 MB

    k1_tables<<<514, 256, 0, stream>>>(Afull, BmT, vvec);
    k2_xBmT <<<1024, 32, 0, stream>>>(x, BmT, T);
    k3_AT   <<<4096, 32, 0, stream>>>(Afull, T, vvec, Wt);
    k4_att  <<<512, 256, 0, stream>>>(Wt, chip, out);
}
